// MemoryBank_33157147525657
// MI455X (gfx1250) — compile-verified, matrix operand order checked by reading, not run on hardware
//
#include <hip/hip_runtime.h>

#define B_ROWS   32768
#define HDIM     256
#define DKEY     64
#define DVAL     128
#define S_SLOTS  8192
#define N_ACT    4096
#define INV_TEMP 10.0f
#define DECAY    0.95f

typedef __attribute__((ext_vector_type(16))) __bf16 v16bf;
typedef __attribute__((ext_vector_type(8)))  float  v8f;
typedef __attribute__((ext_vector_type(2)))  float  v2f;
typedef __attribute__((ext_vector_type(4)))  unsigned int u32x4;
typedef __attribute__((ext_vector_type(4)))  int          i32x4;
typedef __attribute__((ext_vector_type(8)))  int          i32x8;

#if defined(__gfx1250__) && __has_builtin(__builtin_amdgcn_tensor_load_to_lds) && \
    __has_builtin(__builtin_amdgcn_s_wait_tensorcnt)
#define USE_TDM 1
#else
#define USE_TDM 0
#endif

union BFrag { v16bf v; unsigned int u[8]; };

__device__ __forceinline__ unsigned short f2bf(float f) {
  unsigned int u = __builtin_bit_cast(unsigned int, f);
  return (unsigned short)((u + 0x7FFFu + ((u >> 16) & 1u)) >> 16); // RNE truncate
}

// K-base of packed bf16 pair held in fragment dword v for half-lane h
// (ISA 7.12.2: VGPR0-3 hold K 0..7 / 8..15 per half, VGPR4-7 hold K 16..23 / 24..31)
__device__ __forceinline__ int kb_of(int v, int h) {
  return (v < 4 ? 2 * v : 16 + 2 * (v - 4)) + 8 * h;
}

// ---------------------------------------------------------------- utilities
__global__ void zero_mean_kernel(float* __restrict__ mean_raw) {
  int i = blockIdx.x * blockDim.x + threadIdx.x;
  if (i < N_ACT) mean_raw[i] = 0.0f;
}

// gather + L2-normalize active keys -> bf16 kn[NA][64];
// gather values -> bf16 TRANSPOSED vbT[128][NA] (so WMMA B-frags are dword loads)
__global__ void __launch_bounds__(128)
prep_slots_kernel(const float* __restrict__ slots_key,
                  const float* __restrict__ slots_value,
                  const int*   __restrict__ active_idx,
                  unsigned short* __restrict__ kn,
                  unsigned short* __restrict__ vbT) {
  int wave = threadIdx.x >> 5, lane = threadIdx.x & 31;
  int row = blockIdx.x * 4 + wave;
  if (row >= N_ACT) return;
  int idx = active_idx[row];
  const float* kp = slots_key + (long)idx * DKEY;
  float k0 = kp[2 * lane], k1 = kp[2 * lane + 1];
  float ss = k0 * k0 + k1 * k1;
  #pragma unroll
  for (int m = 16; m >= 1; m >>= 1) ss += __shfl_xor(ss, m, 32);
  float inv = 1.0f / fmaxf(sqrtf(ss), 1e-12f);
  unsigned int pk = (unsigned int)f2bf(k0 * inv) | ((unsigned int)f2bf(k1 * inv) << 16);
  ((unsigned int*)kn)[row * (DKEY / 2) + lane] = pk;
  const float* vp = slots_value + (long)idx * DVAL;
  #pragma unroll
  for (int j = 0; j < 4; ++j) {
    int c = lane + 32 * j;
    vbT[(long)c * N_ACT + row] = f2bf(vp[c]);
  }
}

// query = hidden @ key_weight^T via fp32 WMMA (exact-precision output),
// also emits bf16 L2-normalized q into workspace.
__global__ void __launch_bounds__(128)
query_kernel(const float* __restrict__ hidden,
             const float* __restrict__ key_weight,
             float* __restrict__ q_out,
             unsigned short* __restrict__ qn) {
  __shared__ float rowss[16];
  __shared__ float invn[16];
  int wave = threadIdx.x >> 5, lane = threadIdx.x & 31;
  int h = lane >> 4, lm = lane & 15;
  int rowBase = blockIdx.x * 16;
  int col = wave * 16 + lm;                      // 4 waves cover DKEY=64
  if (threadIdx.x < 16) rowss[threadIdx.x] = 0.0f;
  __syncthreads();
  const float* arow = hidden + (long)(rowBase + lm) * HDIM;
  const float* brow = key_weight + (long)col * HDIM;   // B[k][n] = key_weight[n][k]
  v8f c = {};
  for (int kk = 0; kk < HDIM; kk += 4) {
    v2f a, b;
    a.x = arow[kk + 2 * h]; a.y = arow[kk + 2 * h + 1];
    b.x = brow[kk + 2 * h]; b.y = brow[kk + 2 * h + 1];
    c = __builtin_amdgcn_wmma_f32_16x16x4_f32(false, a, false, b, (short)0, c, false, false);
  }
  #pragma unroll
  for (int r = 0; r < 8; ++r) {
    int m = r + 8 * h;
    float q = c[r];
    q_out[(long)(rowBase + m) * DKEY + col] = q;
    atomicAdd(&rowss[m], q * q);
  }
  __syncthreads();
  if (threadIdx.x < 16)
    invn[threadIdx.x] = 1.0f / fmaxf(sqrtf(rowss[threadIdx.x]), 1e-12f);
  __syncthreads();
  #pragma unroll
  for (int r = 0; r < 8; ++r) {
    int m = r + 8 * h;
    qn[(long)(rowBase + m) * DKEY + col] = f2bf(c[r] * invn[m]);
  }
}

// fused: sims (bf16 WMMA) -> softmax (single pass, logits bounded by 1/TEMP)
// -> attn store (non-temporal) -> ctx GEMM (bf16 WMMA) + column sums for mean.
__global__ void __launch_bounds__(256)
attn_kernel(const unsigned short* __restrict__ qn,
            const unsigned short* __restrict__ kn,
            const unsigned short* __restrict__ vbT,
            float* __restrict__ attn_out,
            float* __restrict__ ctx_out,
            float* __restrict__ mean_raw) {
  __shared__ float rowsum[16];
  __shared__ float invsum[16];
  __shared__ float colsum[N_ACT];               // 16 KB
  __shared__ float ctxacc[16 * DVAL];           //  8 KB
  __shared__ unsigned int stage_u[8 * 16 * 16]; //  8 KB: per-wave 16x32 bf16 staging
  __shared__ unsigned int qn_lds_u[16 * DKEY / 2]; // 2 KB: TDM-staged q_norm block
  unsigned short* stage = (unsigned short*)stage_u;

  int wave = threadIdx.x >> 5, lane = threadIdx.x & 31;
  int h = lane >> 4, lm = lane & 15;
  int rowBase = blockIdx.x * 16;

#if USE_TDM
  // ---- TDM: DMA the 16x64 bf16 q_norm block into LDS (issued by wave 0,
  // overlapped with LDS accumulator zeroing, waited via TENSORcnt) ----
  if (wave == 0) {
    unsigned long long ga =
        (unsigned long long)(const void*)(qn + (long)rowBase * DKEY);
    unsigned int lds_off = (unsigned int)(unsigned long long)(void*)qn_lds_u;
    const unsigned int elems = 16 * DKEY; // 1024 x 2-byte elements, 1-D tile
    u32x4 g0 = {1u,                                 // count=1, no gather
                lds_off,                            // D#.lds_addr
                (unsigned int)(ga & 0xFFFFFFFFu),   // global_addr[31:0]
                (unsigned int)((ga >> 32) & 0x01FFFFFFu) | 0x80000000u}; // +type=2
    i32x8 g1 = {0x00010000,                 // workgroup_mask=0, data_size=1 (2B)
                (int)((elems & 0xFFFFu) << 16),     // tensor_dim0[15:0]
                (int)(elems >> 16),                 // tensor_dim0[31:16], tensor_dim1=0
                (int)((elems & 0xFFFFu) << 16),     // tile_dim0 = 1024
                0, 0, 0, 0};                        // tile_dim1/2=0, strides unused (1-D)
    i32x4 z4 = {0, 0, 0, 0};
#if defined(__clang_major__) && (__clang_major__ >= 23)
    i32x8 z8 = {0, 0, 0, 0, 0, 0, 0, 0};
    __builtin_amdgcn_tensor_load_to_lds(g0, g1, z4, z4, z8, 0);
#else
    __builtin_amdgcn_tensor_load_to_lds(g0, g1, z4, z4, 0);
#endif
  }
#endif

  if (threadIdx.x < 16) rowsum[threadIdx.x] = 0.0f;
  for (int j = threadIdx.x; j < N_ACT; j += 256) colsum[j] = 0.0f;
  for (int j = threadIdx.x; j < 16 * DVAL; j += 256) ctxacc[j] = 0.0f;

#if USE_TDM
  if (wave == 0) __builtin_amdgcn_s_wait_tensorcnt(0);
  __syncthreads();
  const unsigned int* qrow = qn_lds_u + lm * (DKEY / 2);   // LDS-resident A rows
#else
  __syncthreads();
  const unsigned int* qrow = (const unsigned int*)(qn + (long)(rowBase + lm) * DKEY);
#endif

  // loop-invariant q_norm A-fragments (K 0..31 and 32..63)
  BFrag a0, a1;
  #pragma unroll
  for (int v = 0; v < 8; ++v) {
    int kb = kb_of(v, h);
    a0.u[v] = qrow[kb >> 1];
    a1.u[v] = qrow[(kb + 32) >> 1];
  }

  // ---------------- pass 1: exp-of-logit row sums ----------------
  for (int i = 0; i < 32; ++i) {
    int t = wave * 32 + i;
    const unsigned short* kbase = kn + (long)(t * 16 + lm) * DKEY;
    if (i + 1 < 32) __builtin_prefetch(kn + (long)((t + 1) * 16 + lm) * DKEY, 0, 1);
    const unsigned int* krow = (const unsigned int*)kbase;
    BFrag b0, b1;
    #pragma unroll
    for (int v = 0; v < 8; ++v) {
      int kb = kb_of(v, h);
      b0.u[v] = krow[kb >> 1];
      b1.u[v] = krow[(kb + 32) >> 1];
    }
    v8f c = {};
    c = __builtin_amdgcn_wmma_f32_16x16x32_bf16(false, a0.v, false, b0.v, (short)0, c, false, false);
    c = __builtin_amdgcn_wmma_f32_16x16x32_bf16(false, a1.v, false, b1.v, (short)0, c, false, false);
    #pragma unroll
    for (int r = 0; r < 8; ++r) {
      float p = __expf(c[r] * INV_TEMP);   // logit <= 10: no max-subtraction needed
      #pragma unroll
      for (int m = 8; m >= 1; m >>= 1) p += __shfl_xor(p, m, 32); // reduce 16 cols
      if (lm == 0) atomicAdd(&rowsum[r + 8 * h], p);
    }
  }
  __syncthreads();
  if (threadIdx.x < 16) invsum[threadIdx.x] = 1.0f / rowsum[threadIdx.x];
  __syncthreads();

  // ---------------- pass 2: recompute, normalize, attn + ctx ----------------
  v8f acc[8];
  v8f zero = {};
  #pragma unroll
  for (int n = 0; n < 8; ++n) acc[n] = zero;
  unsigned short* mystage = stage + wave * (16 * 32);

  for (int i = 0; i < 16; ++i) {
    int chunk = wave * 16 + i;               // K-chunk of 32 slots
    #pragma unroll
    for (int half = 0; half < 2; ++half) {
      int t = 2 * chunk + half;
      const unsigned int* krow = (const unsigned int*)(kn + (long)(t * 16 + lm) * DKEY);
      BFrag b0, b1;
      #pragma unroll
      for (int v = 0; v < 8; ++v) {
        int kb = kb_of(v, h);
        b0.u[v] = krow[kb >> 1];
        b1.u[v] = krow[(kb + 32) >> 1];
      }
      v8f c = {};
      c = __builtin_amdgcn_wmma_f32_16x16x32_bf16(false, a0.v, false, b0.v, (short)0, c, false, false);
      c = __builtin_amdgcn_wmma_f32_16x16x32_bf16(false, a1.v, false, b1.v, (short)0, c, false, false);
      float csum = 0.0f;
      #pragma unroll
      for (int r = 0; r < 8; ++r) {
        int m = r + 8 * h;
        float attn = __expf(c[r] * INV_TEMP) * invsum[m];
        __builtin_nontemporal_store(attn,
            attn_out + (long)(rowBase + m) * N_ACT + t * 16 + lm); // streamed, never re-read
        csum += attn;
        mystage[m * 32 + half * 16 + lm] = f2bf(attn);  // stage for ctx A-fragment
      }
      atomicAdd(&colsum[t * 16 + lm], csum);
    }
    // per-wave cross-lane LDS RAW: DS pipe is in-order per wave; fence compiler+hw
    asm volatile("s_wait_dscnt 0" ::: "memory");
    BFrag af;
    const unsigned int* srow = (const unsigned int*)(mystage + lm * 32);
    #pragma unroll
    for (int v = 0; v < 8; ++v) af.u[v] = srow[kb_of(v, h) >> 1];
    #pragma unroll
    for (int nt = 0; nt < 8; ++nt) {
      const unsigned int* vrow =
          (const unsigned int*)(vbT + (long)(nt * 16 + lm) * N_ACT + chunk * 32);
      BFrag bf_;
      #pragma unroll
      for (int v = 0; v < 8; ++v) bf_.u[v] = vrow[kb_of(v, h) >> 1];
      acc[nt] = __builtin_amdgcn_wmma_f32_16x16x32_bf16(false, af.v, false, bf_.v,
                                                        (short)0, acc[nt], false, false);
    }
  }
  // cross-wave ctx reduction in LDS
  #pragma unroll
  for (int nt = 0; nt < 8; ++nt)
    #pragma unroll
    for (int r = 0; r < 8; ++r)
      atomicAdd(&ctxacc[(r + 8 * h) * DVAL + nt * 16 + lm], acc[nt][r]);
  __syncthreads();
  for (int j = threadIdx.x; j < 16 * DVAL; j += 256)
    ctx_out[(long)rowBase * DVAL + j] = ctxacc[j];
  for (int j = threadIdx.x; j < N_ACT; j += 256)
    atomicAdd(&mean_raw[j], colsum[j]);   // one global atomic per column per WG
}

__global__ void usage_copy_kernel(const float* __restrict__ usage, float* __restrict__ out) {
  int s = blockIdx.x * blockDim.x + threadIdx.x;
  if (s < S_SLOTS) out[s] = usage[s];
}
__global__ void usage_update_kernel(const float* __restrict__ usage,
                                    const int* __restrict__ active,
                                    const float* __restrict__ mean_raw,
                                    float* __restrict__ out) {
  int i = blockIdx.x * blockDim.x + threadIdx.x;
  if (i < N_ACT) {
    int idx = active[i];
    out[idx] = DECAY * usage[idx] +
               (1.0f - DECAY) * (mean_raw[i] * (1.0f / (float)B_ROWS));
  }
}

extern "C" void kernel_launch(void* const* d_in, const int* in_sizes, int n_in,
                              void* d_out, int out_size, void* d_ws, size_t ws_size,
                              hipStream_t stream) {
  const float* hidden      = (const float*)d_in[0];
  const float* key_weight  = (const float*)d_in[1];
  const float* slots_key   = (const float*)d_in[2];
  const float* slots_value = (const float*)d_in[3];
  const float* usage_ema   = (const float*)d_in[4];
  const int*   active_idx  = (const int*)d_in[5];

  float* out     = (float*)d_out;                       // outputs concatenated:
  float* d_ctx   = out;                                 // [B, DVAL]
  float* d_attn  = d_ctx  + (size_t)B_ROWS * DVAL;      // [B, N_ACT]
  float* d_query = d_attn + (size_t)B_ROWS * N_ACT;     // [B, DKEY]
  float* d_usage = d_query + (size_t)B_ROWS * DKEY;     // [S]

  unsigned short* qn  = (unsigned short*)d_ws;          // [B,  DKEY] bf16  (4 MB)
  unsigned short* kn  = qn  + (size_t)B_ROWS * DKEY;    // [NA, DKEY] bf16  (512 KB)
  unsigned short* vbT = kn  + (size_t)N_ACT * DKEY;     // [DVAL, NA] bf16  (1 MB, transposed)
  float* mean_raw = (float*)(vbT + (size_t)DVAL * N_ACT); // [NA] f32 accumulator

  zero_mean_kernel<<<(N_ACT + 255) / 256, 256, 0, stream>>>(mean_raw);
  prep_slots_kernel<<<N_ACT / 4, 128, 0, stream>>>(slots_key, slots_value, active_idx, kn, vbT);
  query_kernel<<<B_ROWS / 16, 128, 0, stream>>>(hidden, key_weight, d_query, qn);
  attn_kernel<<<B_ROWS / 16, 256, 0, stream>>>(qn, kn, vbT, d_attn, d_ctx, mean_raw);
  usage_copy_kernel<<<(S_SLOTS + 255) / 256, 256, 0, stream>>>(usage_ema, d_usage);
  usage_update_kernel<<<(N_ACT + 255) / 256, 256, 0, stream>>>(usage_ema, active_idx, mean_raw, d_usage);
}
